// SwinTransformerBlock_83837761618651
// MI455X (gfx1250) — matrix operand order repaired and checked by reading, not compiled, not run on hardware
//
#include <hip/hip_runtime.h>
#include <hip/hip_bf16.h>
#include <cmath>

typedef __attribute__((ext_vector_type(16))) _Float16 v16h;
typedef __attribute__((ext_vector_type(8)))  float    v8f;

#define WMMA_F16(a, b, c) __builtin_amdgcn_wmma_f32_16x16x32_f16(false, (a), false, (b), (short)0, (c), false, false)

// ---- constants ----
#define Bc   64
#define Hc   56
#define Wc   56
#define Cc   128
#define NHc  4
#define WSc  7
#define SSc  3
#define HIDc 512
#define HDc  32
#define NWc  64           // (56/7)^2
#define Lc   49
#define MTOK 200704       // B*NW*L = B*H*W
#define SCALEc 0.17677669529663687f  // 32^-0.5

// ---------------- fragment loaders (contiguous k order k = 16*(lane/16) + s, same for A and B) ----------------
// A matrix 16x32 f16: lane(0-15)=row M; halves 0..15 -> K=kbase..kbase+15
// where kbase = (lane>=16 ? 16 : 0).
__device__ __forceinline__ v16h frag_a(const _Float16* p0, int ld, int lane, int row0, int k0) {
  const _Float16* p = p0 + (size_t)(row0 + (lane & 15)) * ld + k0 + ((lane >> 4) << 4);
  union { v16h v; float4 q[2]; } u;
  u.q[0] = *(const float4*)(p);
  u.q[1] = *(const float4*)(p + 8);
  return u.v;
}
// B matrix 32x16 f16 (K x N): lane&15 = column N; lanes 0-15 hold K=0..15, lanes 16-31 hold K=16..31.
// Column n of B == row n of a row-major (N,K) weight, so weights load contiguously.
__device__ __forceinline__ v16h frag_b(const _Float16* p0, int ld, int lane, int n0, int k0) {
  const _Float16* p = p0 + (size_t)(n0 + (lane & 15)) * ld + k0 + ((lane >> 4) << 4);
  union { v16h v; float4 q[2]; } u;
  u.q[0] = *(const float4*)(p);
  u.q[1] = *(const float4*)(p + 8);
  return u.v;
}

// ---------------- kernel 0: weights f32 -> f16 ----------------
__global__ void cvt_w_kernel(const float* a, _Float16* oa, int na,
                             const float* b, _Float16* ob, int nb,
                             const float* c, _Float16* oc, int nc,
                             const float* d, _Float16* od, int nd) {
  int i = blockIdx.x * 256 + threadIdx.x;
  if (i < na) oa[i] = (_Float16)a[i];
  if (i < nb) ob[i] = (_Float16)b[i];
  if (i < nc) oc[i] = (_Float16)c[i];
  if (i < nd) od[i] = (_Float16)d[i];
}

// ---------------- kernel 1: LN1 + cyclic shift + window partition -> h16 ----------------
__global__ void ln1_win_kernel(const float* __restrict__ x, const float* __restrict__ w,
                               const float* __restrict__ bch, _Float16* __restrict__ h16) {
  __shared__ float rs[128], rq[128];
  int t = blockIdx.x;                  // windowed token id
  int c = threadIdx.x;                 // channel
  int b  = t / (NWc * Lc);
  int r2 = t % (NWc * Lc);
  int win = r2 / Lc, l = r2 % Lc;
  int hs = (win >> 3) * WSc + l / WSc;
  int ws = (win & 7) * WSc + l % WSc;
  int hi = (hs + SSc) % Hc;            // roll(-SS): shifted[h] = x[(h+SS)%H]
  int wi = (ws + SSc) % Wc;
  float v = x[((size_t)b * (Hc * Wc) + hi * Wc + wi) * Cc + c];
  rs[c] = v; rq[c] = v * v;
  __syncthreads();
  for (int s = 64; s > 0; s >>= 1) {
    if (c < s) { rs[c] += rs[c + s]; rq[c] += rq[c + s]; }
    __syncthreads();
  }
  float mu = rs[0] * (1.f / Cc);
  float var = rq[0] * (1.f / Cc) - mu * mu;
  float rstd = rsqrtf(var + 1e-5f);
  h16[(size_t)t * Cc + c] = (_Float16)((v - mu) * rstd * w[c] + bch[c]);
}

// ---------------- kernel 2: QKV GEMM (WMMA) + head/qkv scatter ----------------
__global__ void qkv_kernel(const _Float16* __restrict__ h16, const _Float16* __restrict__ wq,
                           const float* __restrict__ qb, _Float16* __restrict__ qkv16) {
  int lane = threadIdx.x & 31;
  int wave = threadIdx.x >> 5;
  int mtile = blockIdx.x;              // 16 tokens
  int nt = blockIdx.y * 4 + wave;      // 24 N-tiles over 3C=384
  const _Float16* arow = h16 + (size_t)mtile * 16 * Cc;
  v8f acc = {};
#pragma unroll
  for (int k0 = 0; k0 < Cc; k0 += 32) {
    __builtin_prefetch(wq + (size_t)(nt * 16) * Cc + k0 + 32, 0, 1);
    v16h a = frag_a(arow, Cc, lane, 0, k0);
    v16h b = frag_b(wq, Cc, lane, nt * 16, k0);
    acc = WMMA_F16(a, b, acc);
  }
  int col = nt * 16 + (lane & 15);     // 0..383
  float bias = qb[col];
  int which = col >> 7, head = (col >> 5) & 3, d = col & 31;
#pragma unroll
  for (int r = 0; r < 8; ++r) {
    int t = mtile * 16 + r + ((lane >> 4) << 3);
    int win = t / Lc, l = t % Lc;      // global window id, token in window
    qkv16[(size_t)(((win * NHc + head) * 3 + which) * Lc + l) * HDc + d] =
        (_Float16)(acc[r] + bias);
  }
}

// ---------------- kernel 3: windowed attention (one wave per window-head) ----------------
// LDS layout (single block so asm LDS byte offsets are well-defined; static LDS starts at 0):
//   qs: [0, 4096)  ks: [4096, 8192)  vs: [8192, 12288)  S: [12288, 28672) f32  P: [28672, 36864) f16
#define QS_OFF 0
#define KS_OFF 4096
#define VS_OFF 8192
#define S_OFF  12288
#define P_OFF  28672
__global__ __launch_bounds__(32)
void attn_kernel(const _Float16* __restrict__ qkv16, const float* __restrict__ relb,
                 _Float16* __restrict__ attn16) {
  __shared__ char smem[36864];
  _Float16* qs = (_Float16*)(smem + QS_OFF);
  _Float16* ks = (_Float16*)(smem + KS_OFF);
  _Float16* vs = (_Float16*)(smem + VS_OFF);
  float*    S  = (float*)(smem + S_OFF);
  _Float16* P  = (_Float16*)(smem + P_OFF);
  int lane = threadIdx.x;
  int widx = blockIdx.x >> 2;          // global window 0..4095
  int head = blockIdx.x & 3;
  int win = widx & (NWc - 1);          // window within image
  const _Float16* qg = qkv16 + (size_t)blockIdx.x * 3 * Lc * HDc;
  const _Float16* kg = qg + Lc * HDc;
  const _Float16* vg = kg + Lc * HDc;
  // stage q,k,v rows to LDS via CDNA5 async memory->LDS path (pad rows zero via ds stores)
  for (int s = lane; s < 256; s += 32) {
    int row = s >> 2, seg = (s & 3) << 3;   // seg in halves (8 halves = 16B)
    int eoff = (row * 32 + seg) * 2;        // byte offset within a 4KB tensor slab
    if (row < Lc) {
      const _Float16* gq = qg + row * HDc + seg;
      const _Float16* gk = kg + row * HDc + seg;
      const _Float16* gv = vg + row * HDc + seg;
      unsigned lq = (unsigned)(QS_OFF + eoff);
      unsigned lk = (unsigned)(KS_OFF + eoff);
      unsigned lv = (unsigned)(VS_OFF + eoff);
      asm volatile("global_load_async_to_lds_b128 %0, %1, off" :: "v"(lq), "v"(gq) : "memory");
      asm volatile("global_load_async_to_lds_b128 %0, %1, off" :: "v"(lk), "v"(gk) : "memory");
      asm volatile("global_load_async_to_lds_b128 %0, %1, off" :: "v"(lv), "v"(gv) : "memory");
    } else {
      float4 z; z.x = z.y = z.z = z.w = 0.f;
      *(float4*)&qs[row * 32 + seg] = z;
      *(float4*)&ks[row * 32 + seg] = z;
      *(float4*)&vs[row * 32 + seg] = z;
    }
  }
  asm volatile("s_wait_asynccnt 0" ::: "memory");
  // S = q . k^T  (K=32 == one 16x16x32 WMMA per tile)
#pragma unroll
  for (int mt = 0; mt < 4; ++mt) {
#pragma unroll
    for (int nt = 0; nt < 4; ++nt) {
      v8f acc = {};
      v16h a = frag_a(qs, 32, lane, mt * 16, 0);
      v16h b = frag_b(ks, 32, lane, nt * 16, 0);
      acc = WMMA_F16(a, b, acc);
      int cc = nt * 16 + (lane & 15);
#pragma unroll
      for (int r = 0; r < 8; ++r)
        S[(mt * 16 + r + ((lane >> 4) << 3)) * 64 + cc] = acc[r];
    }
  }
  // softmax rows with inline rel-pos bias + shifted-window mask
  for (int rr = lane; rr < 64; rr += 32) {
    if (rr < Lc) {
      int i1 = rr / WSc, j1 = rr % WSc;
      int hs1 = (win >> 3) * WSc + i1, ws1 = (win & 7) * WSc + j1;
      int reg1 = (hs1 < Hc - WSc ? 0 : (hs1 < Hc - SSc ? 1 : 2)) * 3 +
                 (ws1 < Wc - WSc ? 0 : (ws1 < Wc - SSc ? 1 : 2));
      float mx = -1e30f;
      for (int m = 0; m < Lc; ++m) {
        int i2 = m / WSc, j2 = m % WSc;
        int hs2 = (win >> 3) * WSc + i2, ws2 = (win & 7) * WSc + j2;
        int reg2 = (hs2 < Hc - WSc ? 0 : (hs2 < Hc - SSc ? 1 : 2)) * 3 +
                   (ws2 < Wc - WSc ? 0 : (ws2 < Wc - SSc ? 1 : 2));
        int ridx = (i1 - i2 + WSc - 1) * (2 * WSc - 1) + (j1 - j2 + WSc - 1);
        float v = S[rr * 64 + m] * SCALEc + relb[ridx * NHc + head] +
                  (reg1 == reg2 ? 0.f : -100.f);
        S[rr * 64 + m] = v;
        mx = fmaxf(mx, v);
      }
      float sum = 0.f;
      for (int m = 0; m < Lc; ++m) { float e = __expf(S[rr * 64 + m] - mx); S[rr * 64 + m] = e; sum += e; }
      float inv = 1.f / sum;
      for (int m = 0; m < Lc; ++m) P[rr * 64 + m] = (_Float16)(S[rr * 64 + m] * inv);
      for (int m = Lc; m < 64; ++m) P[rr * 64 + m] = (_Float16)0.f;
    } else {
      for (int m = 0; m < 64; ++m) P[rr * 64 + m] = (_Float16)0.f;
    }
  }
  // out = P @ V   (K=64 -> 2 WMMA K-steps; gather V columns from LDS)
#pragma unroll
  for (int mt = 0; mt < 4; ++mt) {
#pragma unroll
    for (int nt = 0; nt < 2; ++nt) {
      v8f acc = {};
#pragma unroll
      for (int kk = 0; kk < 2; ++kk) {
        v16h a = frag_a(P, 64, lane, mt * 16, kk * 32);
        union { v16h v; _Float16 h[16]; } ub;
        int cn = nt * 16 + (lane & 15);
        int kb = kk * 32 + ((lane >> 4) << 4);
#pragma unroll
        for (int i = 0; i < 16; ++i) ub.h[i] = vs[(kb + i) * 32 + cn];
        acc = WMMA_F16(a, ub.v, acc);
      }
      int cc = nt * 16 + (lane & 15);
#pragma unroll
      for (int r = 0; r < 8; ++r) {
        int row = mt * 16 + r + ((lane >> 4) << 3);
        if (row < Lc)
          attn16[(size_t)(widx * Lc + row) * Cc + head * HDc + cc] = (_Float16)acc[r];
      }
    }
  }
}

// ---------------- kernel 4: proj GEMM + window reverse + unshift + residual ----------------
__global__ void proj_kernel(const _Float16* __restrict__ a16, const _Float16* __restrict__ wp,
                            const float* __restrict__ pb, const float* __restrict__ xin,
                            float* __restrict__ out) {
  int lane = threadIdx.x & 31;
  int wave = threadIdx.x >> 5;
  int mtile = blockIdx.x;
  int nt = blockIdx.y * 4 + wave;      // 8 N-tiles over C=128
  const _Float16* arow = a16 + (size_t)mtile * 16 * Cc;
  v8f acc = {};
#pragma unroll
  for (int k0 = 0; k0 < Cc; k0 += 32) {
    v16h a = frag_a(arow, Cc, lane, 0, k0);
    v16h b = frag_b(wp, Cc, lane, nt * 16, k0);
    acc = WMMA_F16(a, b, acc);
  }
  int col = nt * 16 + (lane & 15);
  float bias = pb[col];
#pragma unroll
  for (int r = 0; r < 8; ++r) {
    int t = mtile * 16 + r + ((lane >> 4) << 3);
    int b_ = t / (NWc * Lc);
    int r2 = t % (NWc * Lc);
    int win = r2 / Lc, l = r2 % Lc;
    int hs = (win >> 3) * WSc + l / WSc;
    int ws = (win & 7) * WSc + l % WSc;
    int ho = (hs + SSc) % Hc;          // reverse roll(+SS)
    int wo = (ws + SSc) % Wc;
    size_t oi = ((size_t)b_ * (Hc * Wc) + ho * Wc + wo) * Cc + col;
    out[oi] = xin[oi] + acc[r] + bias;
  }
}

// ---------------- kernel 5: fused LN2 + FC1 + GELU + FC2 + residual ----------------
__global__ void mlp_kernel(float* __restrict__ xio, const float* __restrict__ n2w,
                           const float* __restrict__ n2b, const _Float16* __restrict__ w1,
                           const float* __restrict__ b1, const _Float16* __restrict__ w2,
                           const float* __restrict__ b2) {
  __shared__ float    xs[16 * 128];
  __shared__ _Float16 y16[16 * 128];
  __shared__ _Float16 hid[16 * 512];
  __shared__ float ps[16 * 8], pq[16 * 8], mus[16], rss[16];
  int tid = threadIdx.x, lane = tid & 31, wave = tid >> 5;
  size_t rowbase = (size_t)blockIdx.x * 16;
  int row = tid >> 3, seg = tid & 7;
  float s = 0.f, sq = 0.f;
  const float* src = xio + (rowbase + row) * Cc + seg * 16;
#pragma unroll
  for (int i = 0; i < 16; ++i) { float v = src[i]; xs[row * 128 + seg * 16 + i] = v; s += v; sq += v * v; }
  ps[row * 8 + seg] = s; pq[row * 8 + seg] = sq;
  __syncthreads();
  if (seg == 0) {
    float S_ = 0.f, Q = 0.f;
    for (int i = 0; i < 8; ++i) { S_ += ps[row * 8 + i]; Q += pq[row * 8 + i]; }
    float mu = S_ * (1.f / Cc);
    mus[row] = mu;
    rss[row] = rsqrtf(Q * (1.f / Cc) - mu * mu + 1e-5f);
  }
  __syncthreads();
  {
    float mu = mus[row], rs = rss[row];
#pragma unroll
    for (int i = 0; i < 16; ++i) {
      int c = seg * 16 + i;
      y16[row * 128 + c] = (_Float16)((xs[row * 128 + c] - mu) * rs * n2w[c] + n2b[c]);
    }
  }
  __syncthreads();
  // FC1 + exact GELU -> hid (16x512) in LDS
  for (int q = 0; q < 8; ++q) {
    int nt = wave * 8 + q;             // 32 N-tiles over HID=512
    v8f acc = {};
#pragma unroll
    for (int k0 = 0; k0 < Cc; k0 += 32) {
      v16h a = frag_a(y16, Cc, lane, 0, k0);
      v16h b = frag_b(w1, Cc, lane, nt * 16, k0);
      acc = WMMA_F16(a, b, acc);
    }
    int col = nt * 16 + (lane & 15);
    float bias = b1[col];
#pragma unroll
    for (int r = 0; r < 8; ++r) {
      int rw = r + ((lane >> 4) << 3);
      float v = acc[r] + bias;
      v = 0.5f * v * (1.f + erff(v * 0.70710678118654752f));
      hid[rw * HIDc + col] = (_Float16)v;
    }
  }
  __syncthreads();
  // FC2 + residual -> xio
  for (int q = 0; q < 2; ++q) {
    int nt = wave * 2 + q;             // 8 N-tiles over C=128
    v8f acc = {};
#pragma unroll
    for (int k0 = 0; k0 < HIDc; k0 += 32) {
      v16h a = frag_a(hid, HIDc, lane, 0, k0);
      v16h b = frag_b(w2, HIDc, lane, nt * 16, k0);
      acc = WMMA_F16(a, b, acc);
    }
    int col = nt * 16 + (lane & 15);
    float bias = b2[col];
#pragma unroll
    for (int r = 0; r < 8; ++r) {
      int rw = r + ((lane >> 4) << 3);
      xio[(rowbase + rw) * Cc + col] = xs[rw * 128 + col] + acc[r] + bias;
    }
  }
}

// ---------------- launch ----------------
extern "C" void kernel_launch(void* const* d_in, const int* in_sizes, int n_in,
                              void* d_out, int out_size, void* d_ws, size_t ws_size,
                              hipStream_t stream) {
  const float* x     = (const float*)d_in[0];
  const float* n1w   = (const float*)d_in[1];
  const float* n1b   = (const float*)d_in[2];
  const float* qkvw  = (const float*)d_in[3];
  const float* qkvb  = (const float*)d_in[4];
  const float* projw = (const float*)d_in[5];
  const float* projb = (const float*)d_in[6];
  const float* relb  = (const float*)d_in[7];
  const float* n2w   = (const float*)d_in[8];
  const float* n2b   = (const float*)d_in[9];
  const float* fc1w  = (const float*)d_in[10];
  const float* fc1b  = (const float*)d_in[11];
  const float* fc2w  = (const float*)d_in[12];
  const float* fc2b  = (const float*)d_in[13];
  float* out = (float*)d_out;

  char* wsb = (char*)d_ws;
  size_t off = 0;
  auto alloc = [&](size_t bytes) -> void* {
    void* p = wsb + off;
    off += (bytes + 255) & ~(size_t)255;
    return p;
  };
  _Float16* wq16  = (_Float16*)alloc((size_t)384 * 128 * 2);
  _Float16* wp16  = (_Float16*)alloc((size_t)128 * 128 * 2);
  _Float16* w116  = (_Float16*)alloc((size_t)512 * 128 * 2);
  _Float16* w216  = (_Float16*)alloc((size_t)128 * 512 * 2);
  _Float16* h16   = (_Float16*)alloc((size_t)MTOK * Cc * 2);
  _Float16* qkv16 = (_Float16*)alloc((size_t)MTOK * 3 * Cc * 2);
  _Float16* a16   = (_Float16*)alloc((size_t)MTOK * Cc * 2);

  cvt_w_kernel<<<256, 256, 0, stream>>>(qkvw, wq16, 384 * 128,
                                        projw, wp16, 128 * 128,
                                        fc1w, w116, 512 * 128,
                                        fc2w, w216, 128 * 512);
  ln1_win_kernel<<<MTOK, 128, 0, stream>>>(x, n1w, n1b, h16);
  qkv_kernel<<<dim3(MTOK / 16, 6), 128, 0, stream>>>(h16, wq16, qkvb, qkv16);
  attn_kernel<<<Bc * NWc * NHc, 32, 0, stream>>>(qkv16, relb, a16);
  proj_kernel<<<dim3(MTOK / 16, 2), 128, 0, stream>>>(a16, wp16, projb, x, out);
  mlp_kernel<<<MTOK / 16, 128, 0, stream>>>(out, n2w, n2b, w116, fc1b, w216, fc2b);
}
